// TaskMoE_57389353009461
// MI455X (gfx1250) — matrix operand order repaired and checked by reading, not compiled, not run on hardware
//
#include <hip/hip_runtime.h>

// Problem constants (from reference)
#define B_ 4
#define L_ 1024
#define D_ 1024
#define H_ 2048
#define E_ 8
#define TOPK_ 2
#define N_ (B_ * L_)   // 4096 tokens

#define MT_ 4          // M tiles per block -> 64 tokens
#define TOKTILE_ 64

typedef __attribute__((ext_vector_type(16))) __bf16 v16bf;
typedef __attribute__((ext_vector_type(8)))  float  v8f;

// ---- helpers -------------------------------------------------------------

__device__ __forceinline__ unsigned short f2bf(float f) {
    // round-to-nearest-even fp32 -> bf16
    unsigned int u = __float_as_uint(f);
    unsigned int r = u + 0x7FFFu + ((u >> 16) & 1u);
    return (unsigned short)(r >> 16);
}

// CDNA5 16-bit A-matrix VGPR layout (ISA 7.12.2, 16-bit A 16x32):
// element j of a lane's v16bf holds K = (j<8 ? j : j+8) + 8*laneHi, M = lane&15.
__device__ __forceinline__ int koff_A(int j, int laneHi) {
    return (j < 8 ? j : j + 8) + (laneHi << 3);
}

// koff_B: contiguous 16-bit B k order (unused; w1t/w2t are packed in koff_A order):
// per 8-VGPR group, lanes 0-15 hold K=0-15, lanes 16-31 hold K=16-31,
// packed sequentially: K = j + 16*laneHi, N = lane&15.
__device__ __forceinline__ int koff_B(int j, int laneHi) {
    return j + (laneHi << 4);
}

__device__ __forceinline__ v8f wmma_bf16(v16bf a, v16bf b, v8f c) {
    return __builtin_amdgcn_wmma_f32_16x16x32_bf16(
        /*neg_a=*/false, a, /*neg_b=*/false, b,
        /*c_mod=*/(short)0, c, /*reuse_a=*/false, /*reuse_b=*/false);
}

// ---- kernel 1: fp32 weights -> bf16 in WMMA-B tiled layout ---------------
// Source W is [E, Kdim, Ndim] row-major. Dest layout:
//   Wt[(((e*KT + kt)*NT + nt)*32 + lane)*16 + j]  (KT=Kdim/32, NT=Ndim/16)
// so the FFN kernel loads one lane's B operand with a single 32-byte load.
__global__ void relayout_w_kernel(const float* __restrict__ W,
                                  unsigned short* __restrict__ Wt,
                                  int Kdim, int Ndim) {
    const int KT = Kdim >> 5, NT = Ndim >> 4;
    long tid = (long)blockIdx.x * blockDim.x + threadIdx.x; // (e, kt, nt, lane)
    int lane = (int)(tid & 31);
    long t = tid >> 5;
    int nt = (int)(t % NT); t /= NT;
    int kt = (int)(t % KT);
    int e  = (int)(t / KT);
    if (e >= E_) return;
    const int laneHi = lane >> 4;
    const int n = nt * 16 + (lane & 15);
    const float* src = W + ((size_t)e * Kdim + (size_t)kt * 32 + (laneHi << 3)) * Ndim + n;
    unsigned short* dst = Wt + ((((size_t)e * KT + kt) * NT + nt) * 32 + lane) * 16;
#pragma unroll
    for (int j = 0; j < 16; ++j) {
        dst[j] = f2bf(src[(size_t)(j < 8 ? j : j + 8) * Ndim]);   // K = (j<8 ? j : j+8) + 8*laneHi (B layout)
    }
}

// ---- kernel 2: gating (logits, softmax, top-2) + x -> bf16 ---------------
// one wave per token
__global__ void gate_kernel(const float* __restrict__ x,
                            const float* __restrict__ Wg,
                            const int* __restrict__ task_p,
                            unsigned short* __restrict__ xb,
                            float* __restrict__ gval,
                            int* __restrict__ lists,
                            int* __restrict__ counts) {
    const int wave = threadIdx.x >> 5;
    const int lane = threadIdx.x & 31;
    const int n = blockIdx.x * 8 + wave;
    if (n >= N_) return;
    const int task = task_p[0];
    const float* xr = x + (size_t)n * D_;
    const float* wg = Wg + (size_t)task * D_ * E_;

    float acc[E_];
#pragma unroll
    for (int e = 0; e < E_; ++e) acc[e] = 0.f;

    for (int i = lane; i < D_; i += 32) {
        float xv = xr[i];
        xb[(size_t)n * D_ + i] = f2bf(xv);
#pragma unroll
        for (int e = 0; e < E_; ++e) acc[e] += xv * wg[(size_t)i * E_ + e];
    }
#pragma unroll
    for (int e = 0; e < E_; ++e) {
#pragma unroll
        for (int off = 16; off > 0; off >>= 1)
            acc[e] += __shfl_xor(acc[e], off, 32);
    }
    if (lane == 0) {
        float m = acc[0];
#pragma unroll
        for (int e = 1; e < E_; ++e) m = fmaxf(m, acc[e]);
        float p[E_]; float s = 0.f;
#pragma unroll
        for (int e = 0; e < E_; ++e) { p[e] = expf(acc[e] - m); s += p[e]; }
        int i1 = 0;
#pragma unroll
        for (int e = 1; e < E_; ++e) if (p[e] > p[i1]) i1 = e;
        int i2 = (i1 == 0) ? 1 : 0;
#pragma unroll
        for (int e = 0; e < E_; ++e) if (e != i1 && p[e] > p[i2]) i2 = e;
        float inv = 1.f / s;
        gval[n * E_ + i1] = p[i1] * inv;
        gval[n * E_ + i2] = p[i2] * inv;
        int s1 = atomicAdd(&counts[i1], 1); lists[i1 * N_ + s1] = n;
        int s2 = atomicAdd(&counts[i2], 1); lists[i2 * N_ + s2] = n;
    }
}

// ---- kernel 3: fused routed 2-layer expert FFN (WMMA bf16) ---------------
// block = (expert e, tile of 64 tokens); 256 threads = 8 waves.
// H is processed in two 1024-wide chunks so LDS fits:
//   xs [32kt][4mt][32lane][16] bf16 (128KB) : x tile, WMMA-A layout, full D
//   hs [32kt][4mt][32lane][16] bf16 (128KB) : gelu(h) chunk, WMMA-A layout
// (kt-major so all 8 A-loads of one k-step sit within DS immediate offsets)
// Layer-2 partials per chunk are combined into y with global f32 atomics.
__global__ __launch_bounds__(256) void moe_ffn_kernel(
    const unsigned short* __restrict__ xb,
    const unsigned short* __restrict__ w1t,
    const unsigned short* __restrict__ w2t,
    const float* __restrict__ b1,
    const float* __restrict__ b2,
    const float* __restrict__ gval,
    const int* __restrict__ lists,
    const int* __restrict__ counts,
    float* __restrict__ y) {

    extern __shared__ char smem[];
    unsigned short* xs = (unsigned short*)smem;            // 32*4*32*16 = 65536 elems
    unsigned short* hs = xs + 32 * MT_ * 32 * 16;          // 65536 elems
    float* gate_s = (float*)(hs + 32 * MT_ * 32 * 16);     // 64 floats
    int*   tok_s  = (int*)(gate_s + TOKTILE_);             // 64 ints

    const int e = blockIdx.y;
    const int tile = blockIdx.x;
    const int cnt = counts[e];
    if (tile * TOKTILE_ >= cnt) return;

    const int tid = threadIdx.x;
    const int wave = tid >> 5;
    const int lane = tid & 31;
    const int laneHi = lane >> 4;

    // stage token ids + gates (pad rows get gate 0 -> contribute nothing)
    if (tid < TOKTILE_) {
        int slot = tile * TOKTILE_ + tid;
        int tok = lists[e * N_ + (slot < cnt ? slot : 0)];
        tok_s[tid] = tok;
        gate_s[tid] = (slot < cnt) ? gval[tok * E_ + e] : 0.f;
    }
    __syncthreads();

    // stage x tile into WMMA-A layout: combos (kt, mt, l2) = 32*4*32 = 4096
    for (int c = tid; c < 32 * MT_ * 32; c += 256) {
        int l2 = c & 31;
        int t2 = c >> 5;
        int mt = t2 & (MT_ - 1);
        int kt = t2 >> 2;
        int row = mt * 16 + (l2 & 15);
        int lHi = l2 >> 4;
        // per-lane A operand = two contiguous 8-element runs of the token's row
        const unsigned short* src =
            xb + (size_t)tok_s[row] * D_ + kt * 32 + (lHi << 3);
        unsigned short* dst = xs + ((size_t)(kt * MT_ + mt) * 32 + l2) * 16;
        *(uint4*)dst       = *(const uint4*)src;          // j=0..7  -> K base
        *(uint4*)(dst + 8) = *(const uint4*)(src + 16);   // j=8..15 -> K base+16
    }
    __syncthreads();

    for (int hc = 0; hc < 2; ++hc) {
        // ---- layer 1: h chunk = gelu(x @ W1[e][:, hc*1024 : +1024] + b1) ----
        // each wave owns 8 chunk-ntiles, 4 passes of (4 mt x 2 nt)
        for (int p = 0; p < 4; ++p) {
            const int nt0 = wave * 8 + p * 2; // chunk-local H ntile
            v8f acc[MT_][2] = {};
#pragma unroll 2
            for (int k = 0; k < 32; ++k) {  // D/32 reduction steps
                const unsigned short* ap = xs + ((size_t)k * MT_ * 32 + lane) * 16;
                v16bf a[MT_];
#pragma unroll
                for (int mt = 0; mt < MT_; ++mt)
                    a[mt] = *(const v16bf*)(ap + (size_t)mt * 32 * 16);
                const unsigned short* bp = w1t +
                    ((((size_t)e * 32 + k) * 128 + (hc * 64 + nt0)) * 32 + lane) * 16;
                v16bf b0  = *(const v16bf*)bp;
                v16bf b1v = *(const v16bf*)(bp + 512);
#pragma unroll
                for (int mt = 0; mt < MT_; ++mt) {
                    acc[mt][0] = wmma_bf16(a[mt], b0,  acc[mt][0]);
                    acc[mt][1] = wmma_bf16(a[mt], b1v, acc[mt][1]);
                }
            }
#pragma unroll
            for (int ntl = 0; ntl < 2; ++ntl) {
                const int colL = (nt0 + ntl) * 16 + (lane & 15); // chunk-local col
                const int col  = hc * 1024 + colL;               // global H col
                const float bias = b1[e * H_ + col];
                // inverse of A-layout mapping for the h store
                const int ktL  = colL >> 5;
                const int koff = colL & 31;
                const int lHi2 = (koff >> 3) & 1;
                const int jj   = (koff & 7) + (koff >= 16 ? 8 : 0);
#pragma unroll
                for (int mt = 0; mt < MT_; ++mt) {
#pragma unroll
                    for (int r = 0; r < 8; ++r) {
                        int row = mt * 16 + (laneHi << 3) + r; // token row 0..63
                        float v = acc[mt][ntl][r] + bias;
                        v = 0.5f * v * (1.0f + erff(v * 0.70710678118f));
                        hs[((size_t)(ktL * MT_ + (row >> 4)) * 32 +
                            (lHi2 * 16 + (row & 15))) * 16 + jj] = f2bf(v);
                    }
                }
            }
        }
        __syncthreads();

        // ---- layer 2 partial: out += h_chunk @ W2[e][hc*1024 : +1024, :] ----
        for (int p = 0; p < 4; ++p) {
            const int nt0 = wave * 8 + p * 2; // D ntile
            v8f acc[MT_][2] = {};
#pragma unroll 2
            for (int k = 0; k < 32; ++k) {
                const unsigned short* ap = hs + ((size_t)k * MT_ * 32 + lane) * 16;
                v16bf a[MT_];
#pragma unroll
                for (int mt = 0; mt < MT_; ++mt)
                    a[mt] = *(const v16bf*)(ap + (size_t)mt * 32 * 16);
                const int kg = hc * 32 + k; // global H ktile
                const unsigned short* bp = w2t +
                    ((((size_t)e * 64 + kg) * 64 + nt0) * 32 + lane) * 16;
                v16bf b0  = *(const v16bf*)bp;
                v16bf b1v = *(const v16bf*)(bp + 512);
#pragma unroll
                for (int mt = 0; mt < MT_; ++mt) {
                    acc[mt][0] = wmma_bf16(a[mt], b0,  acc[mt][0]);
                    acc[mt][1] = wmma_bf16(a[mt], b1v, acc[mt][1]);
                }
            }
#pragma unroll
            for (int ntl = 0; ntl < 2; ++ntl) {
                const int col = (nt0 + ntl) * 16 + (lane & 15); // D column
                const float bias = (hc == 0) ? b2[e * D_ + col] : 0.f;
#pragma unroll
                for (int mt = 0; mt < MT_; ++mt) {
#pragma unroll
                    for (int r = 0; r < 8; ++r) {
                        int row = mt * 16 + (laneHi << 3) + r;
                        float v = (acc[mt][ntl][r] + bias) * gate_s[row];
                        atomicAdd(&y[(size_t)tok_s[row] * D_ + col], v);
                    }
                }
            }
        }
        __syncthreads();
    }
}

// ---- launcher ------------------------------------------------------------

extern "C" void kernel_launch(void* const* d_in, const int* in_sizes, int n_in,
                              void* d_out, int out_size, void* d_ws, size_t ws_size,
                              hipStream_t stream) {
    const float* x   = (const float*)d_in[0];
    const float* Wg  = (const float*)d_in[1];
    const float* W1  = (const float*)d_in[2];
    const float* b1  = (const float*)d_in[3];
    const float* W2  = (const float*)d_in[4];
    const float* b2  = (const float*)d_in[5];
    const int*   tsk = (const int*)d_in[6];
    float* y = (float*)d_out;
    char* ws = (char*)d_ws;

    // workspace layout
    const size_t xb_bytes  = (size_t)N_ * D_ * sizeof(unsigned short);      // 8 MB
    const size_t w1t_bytes = (size_t)E_ * D_ * H_ * sizeof(unsigned short); // 32 MB
    const size_t w2t_bytes = (size_t)E_ * H_ * D_ * sizeof(unsigned short); // 32 MB
    const size_t gv_bytes  = (size_t)N_ * E_ * sizeof(float);
    const size_t li_bytes  = (size_t)E_ * N_ * sizeof(int);

    unsigned short* xb  = (unsigned short*)ws;
    unsigned short* w1t = (unsigned short*)(ws + xb_bytes);
    unsigned short* w2t = (unsigned short*)(ws + xb_bytes + w1t_bytes);
    float* gval  = (float*)(ws + xb_bytes + w1t_bytes + w2t_bytes);
    int*   lists = (int*)(ws + xb_bytes + w1t_bytes + w2t_bytes + gv_bytes);
    int*   cnts  = (int*)(ws + xb_bytes + w1t_bytes + w2t_bytes + gv_bytes + li_bytes);

    hipMemsetAsync(cnts, 0, E_ * sizeof(int), stream);
    hipMemsetAsync(y, 0, (size_t)N_ * D_ * sizeof(float), stream);

    // weight relayout: E*(K/32)*(N/16)*32 threads each doing 16 elems
    relayout_w_kernel<<<(E_ * (D_ / 32) * (H_ / 16) * 32) / 256, 256, 0, stream>>>(
        W1, w1t, D_, H_);
    relayout_w_kernel<<<(E_ * (H_ / 32) * (D_ / 16) * 32) / 256, 256, 0, stream>>>(
        W2, w2t, H_, D_);

    // gating: one wave per token
    gate_kernel<<<N_ / 8, 256, 0, stream>>>(x, Wg, tsk, xb, gval, lists, cnts);

    // fused expert FFN: grid = (token tiles, experts)
    const size_t smem_bytes =
        (size_t)(2 * 32 * MT_ * 32 * 16) * sizeof(unsigned short) +
        TOKTILE_ * sizeof(float) + TOKTILE_ * sizeof(int); // 262,656 B of 320KB LDS
    moe_ffn_kernel<<<dim3(N_ / TOKTILE_, E_), 256, smem_bytes, stream>>>(
        xb, w1t, w2t, b1, b2, gval, lists, cnts, y);
}
